// Coconut_86990267613644
// MI455X (gfx1250) — compile-verified, matrix operand order checked by reading, not run on hardware
//
#include <hip/hip_runtime.h>

// ---------------------------------------------------------------------------
// Problem constants (match reference)
// ---------------------------------------------------------------------------
#define B_    4
#define S_    512
#define D_    512
#define NH_   8
#define DH_   64
#define NL_   2
#define V_    32000
#define L0_   256
#define NLAT_ 8

typedef __attribute__((ext_vector_type(16))) _Float16 v16h;
typedef __attribute__((ext_vector_type(8)))  _Float16 v8h;
typedef __attribute__((ext_vector_type(8)))  float    v8f;
typedef __attribute__((ext_vector_type(4)))  unsigned int u32x4;
typedef __attribute__((ext_vector_type(4)))  int      i32x4;
typedef __attribute__((ext_vector_type(8)))  int      i32x8;

union FragH { v16h v; _Float16 e[16]; };
union FragF { v8f  v; float    e[8];  };

#ifndef ATHENA_TDM
#define ATHENA_TDM 1
#endif
#if ATHENA_TDM && defined(__HIP_DEVICE_COMPILE__) && \
    __has_builtin(__builtin_amdgcn_tensor_load_to_lds)
#define USE_TDM 1
#else
#define USE_TDM 0
#endif

// ---------------------------------------------------------------------------
// Generic batched GEMM: C[z] = alpha * A[z] x B[z] (+ bias), f16 in, f32 acc
//   A: M x K row-major f16 (lda)
//   B: K x N row-major f16 (ldb) if !TRANS_B;  N x K row-major if TRANS_B
//   C: M x N row-major (ldc), f32 or f16 per OUT_HALF
// Per-z offsets: off = (z % zInner)*s?1 + (z / zInner)*s?2
// Block: 256 threads (8 wave32). C tile 64x128, K step 32, 4 WMMAs/wave/step.
// Double-buffered LDS pipeline: K-step i+1 streams in (TDM on the A tile when
// available) while K-step i computes; wave0 defers s_wait_tensorcnt to just
// before the next barrier so the DMA overlaps the whole compute phase.
// ---------------------------------------------------------------------------
template <bool TRANS_B, bool OUT_HALF>
__global__ __launch_bounds__(256) void gemm_wmma(
    const _Float16* __restrict__ A, const _Float16* __restrict__ Bm,
    const float* __restrict__ bias, void* __restrict__ Cv,
    int M, int N, int K, int lda, int ldb, int ldc,
    long long sA1, long long sA2, long long sB1, long long sB2,
    long long sC1, long long sC2, int zInner, float alpha)
{
  __shared__ _Float16 As[2][64 * 32];    // [buf][m][k]
  __shared__ _Float16 Bs[2][128 * 32];   // [buf][n][k]

  const int z  = blockIdx.z;
  const int zi = z % zInner;
  const int zo = z / zInner;
  A  += (long long)zi * sA1 + (long long)zo * sA2;
  Bm += (long long)zi * sB1 + (long long)zo * sB2;
  float*    Cf = (float*)Cv    + (long long)zi * sC1 + (long long)zo * sC2;
  _Float16* Ch = (_Float16*)Cv + (long long)zi * sC1 + (long long)zo * sC2;

  const int row0 = blockIdx.y * 64;
  const int col0 = blockIdx.x * 128;
  const int tid  = threadIdx.x;
  const int lane = tid & 31;
  const int wave = tid >> 5;
  const int wm2  = wave & 1;    // 0..1 : 32-row half of the 64-row tile
  const int wn2  = wave >> 1;   // 0..3 : 32-col quarter of the 128-col tile

  const bool aFull = (row0 + 64 <= M);
  const bool bFull = (col0 + 128 <= N);

  FragF c00, c01, c10, c11;
#pragma unroll
  for (int r = 0; r < 8; ++r) {
    c00.e[r] = 0.0f; c01.e[r] = 0.0f; c10.e[r] = 0.0f; c11.e[r] = 0.0f;
  }

  // Static staging coordinates
  const int am  = tid >> 2;            // 0..63   (A row)
  const int akq = (tid & 3) << 3;      // 0,8,16,24 (A k-chunk of 8 halves)
  const int bnT = tid >> 1;            // TRANS_B: 0..127 (B row = n)
  const int bkT = (tid & 1) << 4;      // TRANS_B: 0/16 (two v8h chunks)
  const int bkN = tid >> 4;            // !TRANS_B: 0..15 (k within half)
  const int bnq = (tid & 15) << 3;     // !TRANS_B: n-chunk of 8

  // ---- Stage A tile (64 x 32 halves) into buffer `buf` for K-offset k0 ----
  auto stageA = [&](int k0, int buf) {
    const bool kFull = (k0 + 32 <= K);
    if (aFull && kFull) {
#if USE_TDM
      if (tid < 32) {
        // Tensor DMA: 2D tile, data_size=2B, tile 32(k) x 64(m), stride lda.
        const unsigned long long gaddr =
            (unsigned long long)(uintptr_t)(A + (long long)row0 * lda + k0);
        const unsigned ldsAddr = (unsigned)(uintptr_t)(&As[buf][0]);
        u32x4 g0;
        g0[0] = 1u;                                        // count=1, user mode
        g0[1] = ldsAddr;                                   // lds_addr (bytes)
        g0[2] = (unsigned)(gaddr & 0xFFFFFFFFull);         // global_addr lo
        g0[3] = (unsigned)((gaddr >> 32) & 0x01FFFFFFull)  // global_addr hi
                | (2u << 30);                              // type=2 ("image")
        i32x8 g1;
        g1[0] = 0x00010000;            // wg_mask=0, data_size=1 (2 bytes)
        g1[1] = (int)(32u << 16);      // tensor_dim0 = 32 (lo16 @ bits63:48)
        g1[2] = (int)(64u << 16);      // dim0 hi=0 | tensor_dim1 = 64 (lo16)
        g1[3] = (int)(32u << 16);      // dim1 hi=0 | tile_dim0 = 32
        g1[4] = (int)(64u);            // tile_dim1 = 64, tile_dim2 = 0
        g1[5] = (int)(unsigned)lda;    // tensor_dim0_stride lo32
        g1[6] = 0;                     // stride hi | dim1_stride lo
        g1[7] = 0;
        const i32x4 z4 = {0, 0, 0, 0};
#if __clang_major__ >= 23
        const i32x8 z8 = {0, 0, 0, 0, 0, 0, 0, 0};
        __builtin_amdgcn_tensor_load_to_lds(g0, g1, z4, z4, z8, 0);
#else
        __builtin_amdgcn_tensor_load_to_lds(g0, g1, z4, z4, 0);
#endif
        // NOTE: no wait here — deferred to the next loop-top barrier.
      }
#else
      const v8h ga = *reinterpret_cast<const v8h*>(
          A + (long long)(row0 + am) * lda + k0 + akq);
      *reinterpret_cast<v8h*>(&As[buf][am * 32 + akq]) = ga;
#endif
    } else {
      for (int i = tid; i < 64 * 32; i += 256) {
        const int m = i >> 5, k = i & 31;
        const int gm = row0 + m, gk = k0 + k;
        const int cm = gm < M - 1 ? gm : M - 1;
        const int ck = gk < K - 1 ? gk : K - 1;
        const float msk = (gm < M && gk < K) ? 1.0f : 0.0f;
        As[buf][i] = (_Float16)((float)A[(long long)cm * lda + ck] * msk);
      }
    }
  };

  // ---- Stage B tile as [n][k] (128 x 32 halves) into buffer `buf` ----
  auto stageB = [&](int k0, int buf) {
    const bool kFull = (k0 + 32 <= K);
    if (TRANS_B) {
      if (bFull && kFull) {
        const _Float16* gp = Bm + (long long)(col0 + bnT) * ldb + k0 + bkT;
        const v8h f0 = *reinterpret_cast<const v8h*>(gp);
        const v8h f1 = *reinterpret_cast<const v8h*>(gp + 8);
        *reinterpret_cast<v8h*>(&Bs[buf][bnT * 32 + bkT]) = f0;
        *reinterpret_cast<v8h*>(&Bs[buf][bnT * 32 + bkT + 8]) = f1;
      } else {
        for (int i = tid; i < 128 * 32; i += 256) {
          const int n = i >> 5, k = i & 31;
          const int gn = col0 + n, gk = k0 + k;
          const int cn = gn < N - 1 ? gn : N - 1;
          const int ck = gk < K - 1 ? gk : K - 1;
          const float msk = (gn < N && gk < K) ? 1.0f : 0.0f;
          Bs[buf][i] = (_Float16)((float)Bm[(long long)cn * ldb + ck] * msk);
        }
      }
    } else {
      if (bFull && kFull) {
#pragma unroll
        for (int seg = 0; seg < 2; ++seg) {
          const int k = bkN + seg * 16;
          const _Float16* gp = Bm + (long long)(k0 + k) * ldb + col0 + bnq;
          union { v8h v; _Float16 e[8]; } pk;
          pk.v = *reinterpret_cast<const v8h*>(gp);
#pragma unroll
          for (int j = 0; j < 8; ++j) Bs[buf][(bnq + j) * 32 + k] = pk.e[j];
        }
      } else {
        for (int i = tid; i < 128 * 32; i += 256) {
          const int n = i >> 5, k = i & 31;
          const int gn = col0 + n, gk = k0 + k;
          const int cn = gn < N - 1 ? gn : N - 1;
          const int ck = gk < K - 1 ? gk : K - 1;
          const float msk = (gn < N && gk < K) ? 1.0f : 0.0f;
          Bs[buf][i] = (_Float16)((float)Bm[(long long)ck * ldb + cn] * msk);
        }
      }
    }
  };

  // ---- Pipelined main loop ----
  stageA(0, 0);
  stageB(0, 0);
  int buf = 0;
  for (int k0 = 0; k0 < K; k0 += 32) {
#if USE_TDM
    if (tid < 32) __builtin_amdgcn_s_wait_tensorcnt(0);
#endif
    __syncthreads();   // current buffer fully staged & prior reads retired
    if (k0 + 32 < K) { // stream next K-step into the other buffer
      stageA(k0 + 32, buf ^ 1);
      stageB(k0 + 32, buf ^ 1);
    }

    // ---- Fragments (ISA 7.12.2 16-bit 16x32 layout):
    // element j of lane l -> k = (j&7) + ((j&8)<<1) + ((l>>4)<<3)
    FragH a0, a1, b0, b1;
    const int halfsel = (lane >> 4) << 3;
    const int rA = (wm2 << 5) + (lane & 15);
    const int nB = (wn2 << 5) + (lane & 15);
#pragma unroll
    for (int j = 0; j < 16; ++j) {
      const int kk = (j & 7) + ((j & 8) << 1) + halfsel;
      a0.e[j] = As[buf][rA * 32 + kk];
      a1.e[j] = As[buf][(rA + 16) * 32 + kk];
      b0.e[j] = Bs[buf][nB * 32 + kk];
      b1.e[j] = Bs[buf][(nB + 16) * 32 + kk];
    }
    c00.v = __builtin_amdgcn_wmma_f32_16x16x32_f16(
        false, a0.v, false, b0.v, (short)0, c00.v, false, false);
    c01.v = __builtin_amdgcn_wmma_f32_16x16x32_f16(
        false, a0.v, false, b1.v, (short)0, c01.v, false, false);
    c10.v = __builtin_amdgcn_wmma_f32_16x16x32_f16(
        false, a1.v, false, b0.v, (short)0, c10.v, false, false);
    c11.v = __builtin_amdgcn_wmma_f32_16x16x32_f16(
        false, a1.v, false, b1.v, (short)0, c11.v, false, false);
    buf ^= 1;
  }

  // ---- Store: 16x16 f32 C layout — VGPR r: lanes0-15 M=r, lanes16-31 M=r+8
  const int mB0 = row0 + (wm2 << 5) + ((lane >> 4) << 3);
  const int cB0 = col0 + (wn2 << 5) + (lane & 15);
#pragma unroll
  for (int r = 0; r < 8; ++r) {
    const int gm0 = mB0 + r;
    const int gm1 = mB0 + 16 + r;
    const int gc0 = cB0;
    const int gc1 = cB0 + 16;
    const float bia0 = (bias && gc0 < N) ? bias[gc0] : 0.0f;
    const float bia1 = (bias && gc1 < N) ? bias[gc1] : 0.0f;
    const float v00 = alpha * c00.e[r] + bia0;
    const float v01 = alpha * c01.e[r] + bia1;
    const float v10 = alpha * c10.e[r] + bia0;
    const float v11 = alpha * c11.e[r] + bia1;
    if (gm0 < M) {
      if (gc0 < N) {
        if (OUT_HALF) Ch[(long long)gm0 * ldc + gc0] = (_Float16)v00;
        else          Cf[(long long)gm0 * ldc + gc0] = v00;
      }
      if (gc1 < N) {
        if (OUT_HALF) Ch[(long long)gm0 * ldc + gc1] = (_Float16)v01;
        else          Cf[(long long)gm0 * ldc + gc1] = v01;
      }
    }
    if (gm1 < M) {
      if (gc0 < N) {
        if (OUT_HALF) Ch[(long long)gm1 * ldc + gc0] = (_Float16)v10;
        else          Cf[(long long)gm1 * ldc + gc0] = v10;
      }
      if (gc1 < N) {
        if (OUT_HALF) Ch[(long long)gm1 * ldc + gc1] = (_Float16)v11;
        else          Cf[(long long)gm1 * ldc + gc1] = v11;
      }
    }
  }
}

// ---------------------------------------------------------------------------
// f32 -> f16 cast (one-time weight conversion)
// ---------------------------------------------------------------------------
__global__ void cast_kernel(const float* __restrict__ in,
                            _Float16* __restrict__ out, long long n)
{
  const long long idx = (long long)blockIdx.x * blockDim.x + threadIdx.x;
  if (idx < n) out[idx] = (_Float16)in[idx];
}

// ---------------------------------------------------------------------------
// LayerNorm (population variance, eps 1e-5): one block per (b,t) row, f16 out
// ---------------------------------------------------------------------------
__global__ __launch_bounds__(256) void ln_kernel(
    const float* __restrict__ x, const float* __restrict__ g,
    const float* __restrict__ b, _Float16* __restrict__ out, int T)
{
  const int row = blockIdx.x;            // 0..B*T-1
  const int bb = row / T, t = row % T;
  const float* xr = x + ((long long)bb * S_ + t) * D_;
  _Float16* orow = out + ((long long)bb * S_ + t) * D_;
  __shared__ float red[256];
  const int tid = threadIdx.x;

  float s = 0.0f;
  for (int i = tid; i < D_; i += 256) s += xr[i];
  red[tid] = s; __syncthreads();
  for (int off = 128; off > 0; off >>= 1) {
    if (tid < off) red[tid] += red[tid + off];
    __syncthreads();
  }
  const float mean = red[0] / (float)D_;
  __syncthreads();

  float sq = 0.0f;
  for (int i = tid; i < D_; i += 256) { float d = xr[i] - mean; sq += d * d; }
  red[tid] = sq; __syncthreads();
  for (int off = 128; off > 0; off >>= 1) {
    if (tid < off) red[tid] += red[tid + off];
    __syncthreads();
  }
  const float inv = rsqrtf(red[0] / (float)D_ + 1e-5f);
  for (int i = tid; i < D_; i += 256)
    orow[i] = (_Float16)((xr[i] - mean) * inv * g[i] + b[i]);
}

// ---------------------------------------------------------------------------
// Causal + padding-mask softmax; f32 scores in, f16 probabilities out
// grid: (T, NH, B); layouts [B, NH, S, S] with leading dim S
// ---------------------------------------------------------------------------
__global__ __launch_bounds__(256) void softmax_kernel(
    const float* __restrict__ scores, _Float16* __restrict__ probs,
    const int* __restrict__ amask, int T)
{
  const int q = blockIdx.x, h = blockIdx.y, bb = blockIdx.z;
  const long long roff = (((long long)bb * NH_ + h) * S_ + q) * S_;
  const float* row = scores + roff;
  _Float16* prow = probs + roff;
  const int* am = amask + (long long)bb * S_;
  const int tid = threadIdx.x;
  __shared__ float red[256];

  float m = -1e30f;
  for (int k = tid; k < T; k += 256) {
    const float v = (k <= q && am[k] > 0) ? row[k] : -1e9f;
    m = fmaxf(m, v);
  }
  red[tid] = m; __syncthreads();
  for (int off = 128; off > 0; off >>= 1) {
    if (tid < off) red[tid] = fmaxf(red[tid], red[tid + off]);
    __syncthreads();
  }
  m = red[0]; __syncthreads();

  float s = 0.0f;
  for (int k = tid; k < T; k += 256) {
    const float v = (k <= q && am[k] > 0) ? row[k] : -1e9f;
    s += __expf(v - m);
  }
  red[tid] = s; __syncthreads();
  for (int off = 128; off > 0; off >>= 1) {
    if (tid < off) red[tid] += red[tid + off];
    __syncthreads();
  }
  const float invs = 1.0f / red[0];
  for (int k = tid; k < T; k += 256) {
    const float v = (k <= q && am[k] > 0) ? row[k] : -1e9f;
    prow[k] = (_Float16)(__expf(v - m) * invs);
  }
}

// ---------------------------------------------------------------------------
// Elementwise kernels
// ---------------------------------------------------------------------------
__global__ void embed_kernel(const int* __restrict__ ids,
                             const float* __restrict__ wte,
                             float* __restrict__ embeds)
{
  const long long idx = (long long)blockIdx.x * blockDim.x + threadIdx.x;
  if (idx >= (long long)B_ * S_ * D_) return;
  const int d = (int)(idx % D_);
  const long long bt = idx / D_;
  embeds[idx] = wte[(long long)ids[bt] * D_ + d];
}

__global__ void addpos_kernel(const float* __restrict__ embeds,
                              const int* __restrict__ pos,
                              const float* __restrict__ wpe,
                              float* __restrict__ x, int T)
{
  const long long idx = (long long)blockIdx.x * blockDim.x + threadIdx.x;
  if (idx >= (long long)B_ * T * D_) return;
  const int d = (int)(idx % D_);
  const long long r = idx / D_;
  const int t = (int)(r % T), bb = (int)(r / T);
  const long long off = ((long long)bb * S_ + t) * D_ + d;
  x[off] = embeds[off] + wpe[(long long)pos[bb * S_ + t] * D_ + d];
}

__global__ void add_kernel(float* __restrict__ x,
                           const float* __restrict__ y, int T)
{
  const long long idx = (long long)blockIdx.x * blockDim.x + threadIdx.x;
  if (idx >= (long long)B_ * T * D_) return;
  const int d = (int)(idx % D_);
  const long long r = idx / D_;
  const int t = (int)(r % T), bb = (int)(r / T);
  const long long off = ((long long)bb * S_ + t) * D_ + d;
  x[off] += y[off];
}

__global__ void gelu_kernel(const float* __restrict__ in,
                            _Float16* __restrict__ out, int T)
{
  const int cols = 4 * D_;
  const long long idx = (long long)blockIdx.x * blockDim.x + threadIdx.x;
  if (idx >= (long long)B_ * T * cols) return;
  const int c = (int)(idx % cols);
  const long long r = idx / cols;
  const int t = (int)(r % T), bb = (int)(r / T);
  const long long off = ((long long)bb * S_ + t) * cols + c;
  const float v = in[off];
  const float u = 0.7978845608028654f * (v + 0.044715f * v * v * v);
  out[off] = (_Float16)(0.5f * v * (1.0f + tanhf(u)));
}

__global__ void copylat_kernel(const _Float16* __restrict__ h,
                               float* __restrict__ embeds, int tok)
{
  const int i = blockIdx.x * blockDim.x + threadIdx.x;
  if (i >= B_ * D_) return;
  const int bb = i / D_, d = i % D_;
  embeds[((long long)bb * S_ + tok) * D_ + d] =
      (float)h[((long long)bb * S_ + (tok - 1)) * D_ + d];
}

__global__ void zero_kernel(float* acc) { acc[0] = 0.0f; acc[1] = 0.0f; }

__global__ __launch_bounds__(256) void loss_row_kernel(
    const float* __restrict__ logits, const int* __restrict__ labels,
    float* __restrict__ acc)
{
  const int row = blockIdx.x;                 // 0..B*(S-1)-1
  const int bb = row / (S_ - 1), t = row % (S_ - 1);
  const int lab = labels[bb * S_ + t + 1];
  if (lab < 0) return;                        // uniform over block
  const float* lr = logits + ((long long)bb * S_ + t) * V_;
  __shared__ float red[256];
  const int tid = threadIdx.x;

  float m = -1e30f;
  for (int v = tid; v < V_; v += 256) m = fmaxf(m, lr[v]);
  red[tid] = m; __syncthreads();
  for (int off = 128; off > 0; off >>= 1) {
    if (tid < off) red[tid] = fmaxf(red[tid], red[tid + off]);
    __syncthreads();
  }
  m = red[0]; __syncthreads();

  float s = 0.0f;
  for (int v = tid; v < V_; v += 256) s += __expf(lr[v] - m);
  red[tid] = s; __syncthreads();
  for (int off = 128; off > 0; off >>= 1) {
    if (tid < off) red[tid] += red[tid + off];
    __syncthreads();
  }
  if (tid == 0) {
    const float nll = (m + __logf(red[0])) - lr[lab];
    atomicAdd(&acc[0], nll);
    atomicAdd(&acc[1], 1.0f);
  }
}

__global__ void finalize_kernel(const float* __restrict__ acc,
                                float* __restrict__ out)
{
  out[0] = acc[0] / fmaxf(acc[1], 1.0f);
}

// ---------------------------------------------------------------------------
// Host side
// ---------------------------------------------------------------------------
struct Blk {
  const float *ln1g, *ln1b, *wqkv, *bqkv, *wo, *bo;
  const float *ln2g, *ln2b, *wfc, *bfc, *wpr, *bpr;
};

static inline long long cdivll(long long a, long long b) { return (a + b - 1) / b; }

extern "C" void kernel_launch(void* const* d_in, const int* in_sizes, int n_in,
                              void* d_out, int out_size, void* d_ws, size_t ws_size,
                              hipStream_t stream)
{
  (void)in_sizes; (void)n_in; (void)out_size; (void)ws_size;

  const int*   ids   = (const int*)d_in[0];
  const int*   amask = (const int*)d_in[1];
  const int*   labels= (const int*)d_in[2];
  const int*   pos   = (const int*)d_in[3];
  const float* wte   = (const float*)d_in[4];
  const float* wpe   = (const float*)d_in[5];

  Blk blk[NL_];
  int ii = 6;
  for (int l = 0; l < NL_; ++l) {
    blk[l].ln1g = (const float*)d_in[ii++]; blk[l].ln1b = (const float*)d_in[ii++];
    blk[l].wqkv = (const float*)d_in[ii++]; blk[l].bqkv = (const float*)d_in[ii++];
    blk[l].wo   = (const float*)d_in[ii++]; blk[l].bo   = (const float*)d_in[ii++];
    blk[l].ln2g = (const float*)d_in[ii++]; blk[l].ln2b = (const float*)d_in[ii++];
    blk[l].wfc  = (const float*)d_in[ii++]; blk[l].bfc  = (const float*)d_in[ii++];
    blk[l].wpr  = (const float*)d_in[ii++]; blk[l].bpr  = (const float*)d_in[ii++];
  }
  const float* lnfg = (const float*)d_in[ii++];
  const float* lnfb = (const float*)d_in[ii++];

  // Workspace layout: f32 region, then f16 region
  const long long BSD  = (long long)B_ * S_ * D_;
  const long long BS3D = (long long)B_ * S_ * 3 * D_;
  const long long BS4D = (long long)B_ * S_ * 4 * D_;
  const long long SCN  = (long long)B_ * NH_ * S_ * S_;
  const long long WQKV = (long long)D_ * 3 * D_;
  const long long WO   = (long long)D_ * D_;
  const long long WFC  = (long long)D_ * 4 * D_;
  const long long WPR  = (long long)4 * D_ * D_;
  const long long WTE  = (long long)V_ * D_;

  float* fws = (float*)d_ws;
  float* embeds = fws;  fws += BSD;
  float* x      = fws;  fws += BSD;
  float* tmp    = fws;  fws += BSD;
  float* fcb    = fws;  fws += BS4D;
  float* scores = fws;  fws += SCN;
  float* acc    = fws;  fws += 2;
  fws += 2;  // keep f16 region 16B-aligned

  _Float16* hws = (_Float16*)fws;
  _Float16* h16    = hws;  hws += BSD;
  _Float16* qkv16  = hws;  hws += BS3D;
  _Float16* attn16 = hws;  hws += BSD;
  _Float16* fcb16  = hws;  hws += BS4D;
  _Float16* probs16= hws;  hws += SCN;
  _Float16* wte16  = hws;  hws += WTE;
  _Float16* w16[NL_][4];
  for (int l = 0; l < NL_; ++l) {
    w16[l][0] = hws; hws += WQKV;   // wqkv
    w16[l][1] = hws; hws += WO;     // wo
    w16[l][2] = hws; hws += WFC;    // wfc
    w16[l][3] = hws; hws += WPR;    // wpr
  }

  float* loss_out = (float*)d_out;
  float* logits   = loss_out + 1;   // [B, S, V]

  auto cast = [&](const float* in, _Float16* out, long long n) {
    cast_kernel<<<(unsigned)cdivll(n, 256), 256, 0, stream>>>(in, out, n);
  };

  auto launch_gemm = [&](bool transB, bool outHalf, const _Float16* A,
                         const _Float16* Bm, const float* bias, void* C,
                         int M, int N, int K, int lda, int ldb, int ldc,
                         long long sA1, long long sA2, long long sB1,
                         long long sB2, long long sC1, long long sC2,
                         int zInner, int Z, float alpha) {
    dim3 grid((N + 127) / 128, (M + 63) / 64, Z);
    if (transB) {
      if (outHalf)
        gemm_wmma<true, true><<<grid, 256, 0, stream>>>(A, Bm, bias, C, M, N,
            K, lda, ldb, ldc, sA1, sA2, sB1, sB2, sC1, sC2, zInner, alpha);
      else
        gemm_wmma<true, false><<<grid, 256, 0, stream>>>(A, Bm, bias, C, M, N,
            K, lda, ldb, ldc, sA1, sA2, sB1, sB2, sC1, sC2, zInner, alpha);
    } else {
      if (outHalf)
        gemm_wmma<false, true><<<grid, 256, 0, stream>>>(A, Bm, bias, C, M, N,
            K, lda, ldb, ldc, sA1, sA2, sB1, sB2, sC1, sC2, zInner, alpha);
      else
        gemm_wmma<false, false><<<grid, 256, 0, stream>>>(A, Bm, bias, C, M, N,
            K, lda, ldb, ldc, sA1, sA2, sB1, sB2, sC1, sC2, zInner, alpha);
    }
  };

  auto run_transformer = [&](int T) {
    const long long nBTD = (long long)B_ * T * D_;
    addpos_kernel<<<(unsigned)cdivll(nBTD, 256), 256, 0, stream>>>(
        embeds, pos, wpe, x, T);
    for (int l = 0; l < NL_; ++l) {
      const Blk& p = blk[l];
      // h16 = LN1(x)
      ln_kernel<<<B_ * T, 256, 0, stream>>>(x, p.ln1g, p.ln1b, h16, T);
      // qkv16 = h16 @ wqkv + b_qkv   (z = batch), f16 out
      launch_gemm(false, true, h16, w16[l][0], p.bqkv, qkv16, T, 3 * D_, D_,
                  D_, 3 * D_, 3 * D_, 0, (long long)S_ * D_, 0, 0, 0,
                  (long long)S_ * 3 * D_, 1, B_, 1.0f);
      // scores = (Q K^T) / sqrt(DH)   (z = b*NH + h), f32 out
      launch_gemm(true, false, qkv16, qkv16 + D_, nullptr, scores, T, T, DH_,
                  3 * D_, 3 * D_, S_, DH_, (long long)S_ * 3 * D_, DH_,
                  (long long)S_ * 3 * D_, (long long)S_ * S_,
                  (long long)NH_ * S_ * S_, NH_, B_ * NH_, 0.125f);
      dim3 sg(T, NH_, B_);
      softmax_kernel<<<sg, 256, 0, stream>>>(scores, probs16, amask, T);
      // attn16 = P @ V, f16 out
      launch_gemm(false, true, probs16, qkv16 + 2 * D_, nullptr, attn16, T,
                  DH_, T, S_, 3 * D_, D_, (long long)S_ * S_,
                  (long long)NH_ * S_ * S_, DH_, (long long)S_ * 3 * D_, DH_,
                  (long long)S_ * D_, NH_, B_ * NH_, 1.0f);
      // x += attn16 @ w_o + b_o
      launch_gemm(false, false, attn16, w16[l][1], p.bo, tmp, T, D_, D_, D_,
                  D_, D_, 0, (long long)S_ * D_, 0, 0, 0, (long long)S_ * D_,
                  1, B_, 1.0f);
      add_kernel<<<(unsigned)cdivll(nBTD, 256), 256, 0, stream>>>(x, tmp, T);
      // h16 = LN2(x)
      ln_kernel<<<B_ * T, 256, 0, stream>>>(x, p.ln2g, p.ln2b, h16, T);
      // fcb = h16 @ w_fc + b_fc ; fcb16 = gelu(fcb)
      launch_gemm(false, false, h16, w16[l][2], p.bfc, fcb, T, 4 * D_, D_, D_,
                  4 * D_, 4 * D_, 0, (long long)S_ * D_, 0, 0, 0,
                  (long long)S_ * 4 * D_, 1, B_, 1.0f);
      gelu_kernel<<<(unsigned)cdivll((long long)B_ * T * 4 * D_, 256), 256, 0,
                    stream>>>(fcb, fcb16, T);
      // x += fcb16 @ w_pr + b_pr
      launch_gemm(false, false, fcb16, w16[l][3], p.bpr, tmp, T, D_, 4 * D_,
                  4 * D_, D_, D_, 0, (long long)S_ * 4 * D_, 0, 0, 0,
                  (long long)S_ * D_, 1, B_, 1.0f);
      add_kernel<<<(unsigned)cdivll(nBTD, 256), 256, 0, stream>>>(x, tmp, T);
    }
    // h16 = LN_f(x)
    ln_kernel<<<B_ * T, 256, 0, stream>>>(x, lnfg, lnfb, h16, T);
  };

  // ----- one-time f16 weight conversion (per call; L2-resident afterwards)
  for (int l = 0; l < NL_; ++l) {
    cast(blk[l].wqkv, w16[l][0], WQKV);
    cast(blk[l].wo,   w16[l][1], WO);
    cast(blk[l].wfc,  w16[l][2], WFC);
    cast(blk[l].wpr,  w16[l][3], WPR);
  }
  cast(wte, wte16, WTE);

  // ----- full forward with latent feedback -----
  embed_kernel<<<(unsigned)cdivll(BSD, 256), 256, 0, stream>>>(ids, wte, embeds);
  zero_kernel<<<1, 1, 0, stream>>>(acc);

  int start = 0;
  for (int p = 0; p <= NLAT_; ++p) {
    const int T = (p < NLAT_) ? (L0_ + p) : S_;
    run_transformer(T);
    const int M = T - start;
    // logits[:, start:T] = h16[:, start:T] @ wte^T   (z = batch)
    launch_gemm(true, false, h16 + (long long)start * D_, wte16, nullptr,
                logits + (long long)start * V_, M, V_, D_, D_, D_, V_, 0,
                (long long)S_ * D_, 0, 0, 0, (long long)S_ * V_, 1, B_, 1.0f);
    if (p < NLAT_)
      copylat_kernel<<<(B_ * D_ + 255) / 256, 256, 0, stream>>>(h16, embeds, T);
    start = T;
  }

  loss_row_kernel<<<B_ * (S_ - 1), 256, 0, stream>>>(logits, labels, acc);
  finalize_kernel<<<1, 1, 0, stream>>>(acc, loss_out);
}
